// DeepGCN_3453153706770
// MI455X (gfx1250) — matrix operand order repaired and checked by reading, not compiled, not run on hardware
//
#include <hip/hip_runtime.h>
#include <hip/hip_bf16.h>

typedef __attribute__((ext_vector_type(16))) _Float16 v16h;
typedef __attribute__((ext_vector_type(8)))  _Float16 v8h;
typedef __attribute__((ext_vector_type(8)))  float    v8f;

#if __has_builtin(__builtin_amdgcn_sched_barrier)
#define SCHED_FENCE() __builtin_amdgcn_sched_barrier(0)
#else
#define SCHED_FENCE()
#endif

// ---------------------------------------------------------------- utilities

__global__ void k_fill1(float* __restrict__ p, int n) {
  int i = blockIdx.x * 256 + threadIdx.x;
  if (i < n) p[i] = 1.0f;  // self-loop contribution to degree
}

__global__ void k_count(const int* __restrict__ row, float* __restrict__ deg, int E) {
  int i = blockIdx.x * 256 + threadIdx.x;
  if (i < E) atomicAdd(&deg[row[i]], 1.0f);
}

__global__ void k_rsqrt(float* __restrict__ p, int n) {
  int i = blockIdx.x * 256 + threadIdx.x;
  if (i < n) p[i] = rsqrtf(p[i]);  // deg -> dinv in place (deg >= 1 always)
}

__global__ void k_norm(const int* __restrict__ row, const int* __restrict__ col,
                       const float* __restrict__ dinv, float* __restrict__ nrm, int E) {
  int i = blockIdx.x * 256 + threadIdx.x;
  if (i < E) nrm[i] = dinv[row[i]] * dinv[col[i]];
}

// vectorized f32 -> f16 (8 elems/thread: b128 loads -> b128 store)
__global__ void k_f32_to_f16v(const float* __restrict__ s, _Float16* __restrict__ d, int n8) {
  int i = blockIdx.x * 256 + threadIdx.x;
  if (i >= n8) return;
  v8f v = reinterpret_cast<const v8f*>(s)[i];
  reinterpret_cast<v8h*>(d)[i] = __builtin_convertvector(v, v8h);
}

// vectorized relu + f32 -> f16
__global__ void k_relu_f16v(const float* __restrict__ s, _Float16* __restrict__ d, int n8) {
  int i = blockIdx.x * 256 + threadIdx.x;
  if (i >= n8) return;
  v8f v = reinterpret_cast<const v8f*>(s)[i];
  v8f z = {};
  v = __builtin_elementwise_max(v, z);
  reinterpret_cast<v8h*>(d)[i] = __builtin_convertvector(v, v8h);
}

__global__ void k_pad_w(const float* __restrict__ w, _Float16* __restrict__ d,
                        int K, int C, int CP) {
  int i = blockIdx.x * 256 + threadIdx.x;
  if (i >= K * CP) return;
  int k = i / CP, c = i - k * CP;
  d[i] = (c < C) ? (_Float16)w[k * C + c] : (_Float16)0.0f;
}

// ------------------------------------------- WMMA f16 GEMM + fused agg-init
// A: [M, K] f16 row-major, B: [K, NOUT] f16 row-major.
// Outputs: C[m][c]   = (A@B)[m][c]                       (raw, for edge gather)
//          agg[m][c] = bias[c] + C[m][c] * dinv[m]^2     (bias + self-loop seed)
// 256 threads = 8 waves; each wave owns a 16-row strip, full NOUT columns.
// B staged 128 K-rows per barrier, transposed in LDS: Bs[col][k] (k contiguous),
// so each lane reads its WMMA B-fragment as two contiguous ds_load_b128.
// sched_barrier fences force load-clause / wmma-chain separation so one wait
// covers a whole fragment clause instead of one wait per WMMA.
// Requires K % 128 == 0 (holds: K = 512 / 128).
template <int NOUT, int WAGG>
__global__ __launch_bounds__(256) void k_gemm_wmma(
    const _Float16* __restrict__ A, const _Float16* __restrict__ B,
    float* __restrict__ C, const float* __restrict__ bias,
    const float* __restrict__ dinv, float* __restrict__ agg, int M, int K) {
  constexpr int NT = NOUT / 16;
  constexpr int KC = 128;                    // K-chunk staged per barrier
  constexpr int KS = KC / 32;                // k-steps per chunk
  __shared__ unsigned short Bs[NOUT * KC];   // [col][k]

  const int tid  = threadIdx.x;
  const int wave = tid >> 5;
  const int lane = tid & 31;
  const int rowBase = blockIdx.x * 128 + wave * 16;
  const int mrow  = rowBase + (lane & 15);
  const int msafe = (mrow < M) ? mrow : 0;   // keep loads in range; stores guarded below
  const int kh    = lane >> 4;               // lane-group selector
  const int n0    = lane & 15;

  v8f acc[NT];
#pragma unroll
  for (int t = 0; t < NT; ++t) acc[t] = (v8f){0.f, 0.f, 0.f, 0.f, 0.f, 0.f, 0.f, 0.f};

  const ushort4* Bv = reinterpret_cast<const ushort4*>(B);  // 4 halves per load
  const _Float16* Arow = A + (size_t)msafe * K;

  for (int k0 = 0; k0 < K; k0 += KC) {
    // prefetch next A chunk (global_prefetch_b8)
    if (k0 + KC < K) __builtin_prefetch(Arow + k0 + KC + kh * 8, 0, 0);

    // ---- stage B[k0:k0+KC, :] transposed into Bs[col][k]
    constexpr int NQ = (KC * NOUT) / (256 * 4);
    ushort4 qbuf[NQ];
#pragma unroll
    for (int i = 0; i < NQ; ++i) {
      int e = tid + i * 256;  // quad index
      int k = e / (NOUT / 4);
      int c4 = e % (NOUT / 4);
      qbuf[i] = Bv[(size_t)(k0 + k) * (NOUT / 4) + c4];
    }
    SCHED_FENCE();  // keep all global loads as one clause before the ds stores
#pragma unroll
    for (int i = 0; i < NQ; ++i) {
      int e = tid + i * 256;
      int k = e / (NOUT / 4);
      int c4 = e % (NOUT / 4);
      Bs[(c4 * 4 + 0) * KC + k] = qbuf[i].x;
      Bs[(c4 * 4 + 1) * KC + k] = qbuf[i].y;
      Bs[(c4 * 4 + 2) * KC + k] = qbuf[i].z;
      Bs[(c4 * 4 + 3) * KC + k] = qbuf[i].w;
    }
    __syncthreads();

    // ---- preload all A fragments for this chunk (one global clause)
    // 16-bit A 16x32 layout: lane(0..15)=row M, kh selects K-group;
    // halves 0..7 = K[kh*8 .. +7], halves 8..15 = K[16+kh*8 .. +7]
    v16h afrag[KS];
#pragma unroll
    for (int s = 0; s < KS; ++s) {
      const _Float16* ap = Arow + k0 + s * 32 + kh * 8;
      v8h alo = *reinterpret_cast<const v8h*>(ap);
      v8h ahi = *reinterpret_cast<const v8h*>(ap + 16);
      afrag[s] = __builtin_shufflevector(alo, ahi, 0, 1, 2, 3, 4, 5, 6, 7,
                                         8, 9, 10, 11, 12, 13, 14, 15);
    }
    SCHED_FENCE();

#pragma unroll
    for (int s = 0; s < KS; ++s) {
      const int kk = s * 32;
      // preload ALL B fragments for this k-step (one ds clause, one wait)
      v16h bfrag[NT];
#pragma unroll
      for (int t = 0; t < NT; ++t) {
        // B fragment: lane(0..15)=col N; halves 0..7 = K[kk+kh*8 .. +7], halves 8..15 = K[kk+16+kh*8 .. +7]
        const unsigned short* bp = &Bs[(t * 16 + n0) * KC + kk + kh * 8];
        v8h blo = *reinterpret_cast<const v8h*>(bp);
        v8h bhi = *reinterpret_cast<const v8h*>(bp + 16);
        bfrag[t] = __builtin_shufflevector(blo, bhi, 0, 1, 2, 3, 4, 5, 6, 7,
                                           8, 9, 10, 11, 12, 13, 14, 15);
      }
      SCHED_FENCE();  // ds clause above, back-to-back wmma chain below
#pragma unroll
      for (int t = 0; t < NT; ++t) {
        acc[t] = __builtin_amdgcn_wmma_f32_16x16x32_f16(
            /*neg_a=*/false, afrag[s], /*neg_b=*/false, bfrag[t],
            /*c_mod=*/(short)0, acc[t], /*reuse_a=*/false, /*reuse_b=*/false);
      }
      SCHED_FENCE();
    }
    __syncthreads();
  }

  // C/D layout: VGPR r, lanes 0-15 -> M=r, lanes 16-31 -> M=8+r; N = lane&15
  const int rbase = (lane >> 4) * 8;
  if (rowBase + 16 <= M) {
    // full tile: wave-uniform guard, unconditional stores
    float dv2[8];
#pragma unroll
    for (int r = 0; r < 8; ++r) {
      float d = dinv[rowBase + rbase + r];
      dv2[r] = d * d;
    }
#pragma unroll
    for (int t = 0; t < NT; ++t) {
      int c = t * 16 + n0;
      float bc = (WAGG == NOUT || c < WAGG) ? bias[(WAGG == NOUT) ? c : (c < WAGG ? c : 0)] : 0.f;
#pragma unroll
      for (int r = 0; r < 8; ++r) {
        int m = rowBase + rbase + r;
        float val = acc[t][r];
        C[(size_t)m * NOUT + c] = val;
        if (WAGG == NOUT || c < WAGG)
          agg[(size_t)m * WAGG + c] = bc + val * dv2[r];
      }
    }
  } else if (rowBase < M) {
    // partial tile (only if M % 16 != 0)
#pragma unroll
    for (int t = 0; t < NT; ++t) {
      int c = t * 16 + n0;
#pragma unroll
      for (int r = 0; r < 8; ++r) {
        int m = rowBase + rbase + r;
        if (m < M) {
          float val = acc[t][r];
          C[(size_t)m * NOUT + c] = val;
          if (c < WAGG) {
            float d = dinv[m];
            agg[(size_t)m * WAGG + c] = bias[c] + val * d * d;
          }
        }
      }
    }
  }
}

// ----------------------------------------------------------- aggregation

// agg[row[e]][:] += norm[e] * h[col[e]][:]  — one thread per (edge, float4 chunk)
template <int WIDTH>
__global__ void k_scatter(const int* __restrict__ row, const int* __restrict__ col,
                          const float* __restrict__ nrm, const float* __restrict__ h,
                          int hstride, float* __restrict__ agg, int E) {
  constexpr int CH = WIDTH / 4;
  int idx = blockIdx.x * 256 + threadIdx.x;
  if (idx >= E * CH) return;
  int e = idx / CH, q = idx - e * CH;
  int r = row[e], c = col[e];
  float nm = nrm[e];
  float4 hv = *reinterpret_cast<const float4*>(h + (size_t)c * hstride + q * 4);
  float* ap = agg + (size_t)r * WIDTH + q * 4;
  atomicAdd(ap + 0, hv.x * nm);
  atomicAdd(ap + 1, hv.y * nm);
  atomicAdd(ap + 2, hv.z * nm);
  atomicAdd(ap + 3, hv.w * nm);
}

// --------------------------------------------------------- log_softmax (C<=64)
__global__ void k_logsoftmax(const float* __restrict__ agg, float* __restrict__ out,
                             int N, int C) {
  int gwarp = (blockIdx.x * blockDim.x + threadIdx.x) >> 5;
  int lane  = threadIdx.x & 31;
  if (gwarp >= N) return;  // uniform per wave
  const float* rp = agg + (size_t)gwarp * C;
  float v0 = (lane < C)      ? rp[lane]      : -__builtin_inff();
  float v1 = (lane + 32 < C) ? rp[lane + 32] : -__builtin_inff();
  float mx = fmaxf(v0, v1);
#pragma unroll
  for (int off = 16; off > 0; off >>= 1) mx = fmaxf(mx, __shfl_xor(mx, off, 32));
  float s = ((lane < C) ? __expf(v0 - mx) : 0.f) +
            ((lane + 32 < C) ? __expf(v1 - mx) : 0.f);
#pragma unroll
  for (int off = 16; off > 0; off >>= 1) s += __shfl_xor(s, off, 32);
  float lse = mx + __logf(s);
  if (lane < C)      out[(size_t)gwarp * C + lane]      = v0 - lse;
  if (lane + 32 < C) out[(size_t)gwarp * C + lane + 32] = v1 - lse;
}

// ----------------------------------------------------------------- launch

extern "C" void kernel_launch(void* const* d_in, const int* in_sizes, int n_in,
                              void* d_out, int out_size, void* d_ws, size_t ws_size,
                              hipStream_t stream) {
  const float* x  = (const float*)d_in[0];
  const int*   ei = (const int*)d_in[1];
  const float* W1 = (const float*)d_in[2];
  const float* b1 = (const float*)d_in[3];
  const float* W2 = (const float*)d_in[4];
  const float* b2 = (const float*)d_in[5];
  const float* W3 = (const float*)d_in[6];
  const float* b3 = (const float*)d_in[7];
  float* out = (float*)d_out;

  const int HID = in_sizes[3];        // 128
  const int C   = in_sizes[7];        // 40
  const int FIN = in_sizes[2] / HID;  // 512
  const int N   = in_sizes[0] / FIN;  // 50000
  const int E   = in_sizes[1] / 2;    // 800000
  const int CP  = 64;                 // padded class count for WMMA GEMM 3

  char* base = (char*)d_ws;
  size_t o = 0;
  auto take = [&](size_t bytes) {
    char* p = base + o;
    o += (bytes + 255) & ~(size_t)255;
    return p;
  };
  float*    dinv = (float*)take((size_t)N * 4);             // deg -> dinv
  float*    nrm  = (float*)take((size_t)E * 4);
  _Float16* x16  = (_Float16*)take((size_t)N * FIN * 2);
  _Float16* w16  = (_Float16*)take((size_t)FIN * HID * 2);  // reused per layer
  float*    h    = (float*)take((size_t)N * HID * 4);       // GEMM out (f32)
  float*    agg  = (float*)take((size_t)N * HID * 4);       // aggregation out
  _Float16* h16  = (_Float16*)take((size_t)N * HID * 2);    // next-layer input
  (void)ws_size; (void)n_in; (void)out_size;

  const int* erow = ei;      // targets
  const int* ecol = ei + E;  // sources

  auto blk = [](long long n) { return (int)((n + 255) / 256); };
  const int gemmGrid = (N + 127) / 128;

  // ---- symmetric normalization  D^-1/2 (A+I) D^-1/2
  k_fill1 <<<blk(N), 256, 0, stream>>>(dinv, N);
  k_count <<<blk(E), 256, 0, stream>>>(erow, dinv, E);
  k_rsqrt <<<blk(N), 256, 0, stream>>>(dinv, N);
  k_norm  <<<blk(E), 256, 0, stream>>>(erow, ecol, dinv, nrm, E);

  // ---- layer 1: agg = b1 + selfloop(x@W1); agg += edges; h16 = relu(agg)
  k_f32_to_f16v<<<blk((long long)N * FIN / 8), 256, 0, stream>>>(x, x16, N / 8 * FIN);
  k_f32_to_f16v<<<blk(FIN * HID / 8), 256, 0, stream>>>(W1, w16, FIN * HID / 8);
  k_gemm_wmma<128, 128><<<gemmGrid, 256, 0, stream>>>(x16, w16, h, b1, dinv, agg, N, FIN);
  k_scatter<128><<<blk((long long)E * (128 / 4)), 256, 0, stream>>>(erow, ecol, nrm, h, HID, agg, E);
  k_relu_f16v<<<blk((long long)N * HID / 8), 256, 0, stream>>>(agg, h16, N * HID / 8);

  // ---- layer 2
  k_f32_to_f16v<<<blk(HID * HID / 8), 256, 0, stream>>>(W2, w16, HID * HID / 8);
  k_gemm_wmma<128, 128><<<gemmGrid, 256, 0, stream>>>(h16, w16, h, b2, dinv, agg, N, HID);
  k_scatter<128><<<blk((long long)E * (128 / 4)), 256, 0, stream>>>(erow, ecol, nrm, h, HID, agg, E);
  k_relu_f16v<<<blk((long long)N * HID / 8), 256, 0, stream>>>(agg, h16, N * HID / 8);

  // ---- layer 3 (pad 40 -> 64 cols for WMMA) + log_softmax
  k_pad_w<<<blk(HID * CP), 256, 0, stream>>>(W3, w16, HID, C, CP);
  k_gemm_wmma<64, 40><<<gemmGrid, 256, 0, stream>>>(h16, w16, h, b3, dinv, agg, N, HID);
  k_scatter<40><<<blk((long long)E * (40 / 4)), 256, 0, stream>>>(erow, ecol, nrm, h, CP, agg, E);
  k_logsoftmax<<<(N + 7) / 8, 256, 0, stream>>>(agg, out, N, C);
}
